// SPD_11888469475438
// MI455X (gfx1250) — matrix operand order repaired and checked by reading, not compiled, not run on hardware
//
#include <hip/hip_runtime.h>
#include <math.h>
#include <stdint.h>

// ---------------------------------------------------------------------------
// SPD (SnowflakeNet point-splitting decoder) forward for MI455X / gfx1250.
// All dense layers run through v_wmma_f32_16x16x32_f16 (f16 in, f32 accum).
// Intermediates use channel-major (C, B*N) layout; attention section is
// processed in 8 column chunks of 32768 (BNK = 262144) to keep workspace
// bounded (~270 MB) and chunk buffers L2-resident.
// rowmax uses GLOBAL_LOAD_ASYNC_TO_LDS_B128 (+ s_wait_asynccnt) staging.
// ---------------------------------------------------------------------------

typedef __attribute__((ext_vector_type(16))) _Float16 v16h;
typedef __attribute__((ext_vector_type(8)))  float    v8f;
typedef __attribute__((ext_vector_type(8)))  unsigned v8u;

#define BB    8
#define NN    2048
#define KNNK  16
#define DIMF  2
#define UPF   2
#define MM    (BB*NN)        // 16384 columns (B*N)
#define NN2   (NN*UPF)       // 4096
#define MM2   (BB*NN2)       // 32768 columns (B*N*UP)
#define BNK   (MM*KNNK)      // 262144 columns (B*N*K)
#define CHCOL 32768          // chunk columns for attention section
#define NCHNK (BNK/CHCOL)    // 8
#define CHPTS (CHCOL/KNNK)   // 2048 points per chunk

#define FLAG_RELU     1
#define FLAG_TANH     2
#define FLAG_PRE_RELU 4
#define FLAG_RES      8

#define TC 256   // output columns per block (8 waves x 2 tiles x 16)
#define KC 32    // K chunk per WMMA step

static __device__ inline unsigned pack2h(float a, float b) {
    union { _Float16 h[2]; unsigned u; } p;
    p.h[0] = (_Float16)a; p.h[1] = (_Float16)b;
    return p.u;
}

// ---------------------------------------------------------------------------
// Generic WMMA GEMM:  out[o, col] = act( sum_c W[o,c]*actX(X[c*x_ld+col]) + b[o] (+res) )
// out index = o*out_ld + col*cs + coff   (cs/coff support strided deconv writes)
// LDS layouts:
//   wt: f16 row-major [m][k]            (A tile 16 x 32)
//   xt: u32 pair-interleaved [k/2][col] (B tile 32 x 256; word = halves {k,k+1})
// ---------------------------------------------------------------------------
__global__ __launch_bounds__(256)
void wmma_gemm_kernel(const float* __restrict__ Wm, const float* __restrict__ X,
                      const float* __restrict__ bias, const float* __restrict__ res,
                      float* __restrict__ out,
                      int O, int C, int M, int x_ld, int out_ld,
                      int cs, int coff, int flags)
{
    __shared__ unsigned wtu[16 * KC / 2];        // 16x32 f16 (u32 view)
    __shared__ unsigned xtu[(KC / 2) * TC];      // 32x256 f16 pair-interleaved

    _Float16* wth = (_Float16*)wtu;
    _Float16* xth = (_Float16*)xtu;

    const int tid  = threadIdx.x;
    const int lane = tid & 31;              // wave32
    const int wave = tid >> 5;              // 0..7
    const int colBlock = blockIdx.x * TC;
    const int oTile    = blockIdx.y * 16;
    const bool fullO = (oTile + 16) <= O;
    const bool pre   = (flags & FLAG_PRE_RELU) != 0;

    v8f acc0 = {}, acc1 = {};
    const int nK = (C + KC - 1) / KC;

    for (int kc = 0; kc < nK; ++kc) {
        const int k0 = kc * KC;
        const bool fullK = (k0 + KC) <= C;

        // ---- stage W tile (16 x 32) ----
        if (fullK && fullO && (C % 4) == 0) {
            if (tid < 128) {
                const int m = tid >> 3, kq = (tid & 7) * 4;
                const float4 w4 = *(const float4*)(Wm + (size_t)(oTile + m) * C + k0 + kq);
                uint2 pw; pw.x = pack2h(w4.x, w4.y); pw.y = pack2h(w4.z, w4.w);
                *(uint2*)&wtu[m * (KC / 2) + (kq >> 1)] = pw;
            }
        } else {
            for (int i = tid; i < 16 * KC; i += 256) {
                const int m = i / KC, k = i % KC;
                const int o = oTile + m, kk = k0 + k;
                float v = 0.0f;
                if (o < O && kk < C) v = Wm[(size_t)o * C + kk];
                wth[i] = (_Float16)v;
            }
        }

        // ---- stage X tile (32 x 256), pair-interleaved ----
        if (fullK) {
            #pragma unroll
            for (int it = 0; it < 4; ++it) {
                const int e  = tid + it * 256;       // 0..1023 quad units
                const int kp = e >> 6;               // 0..15 (k pair)
                const int cq = (e & 63) * 4;         // 0..252
                const float* p0 = X + (size_t)(k0 + 2 * kp    ) * x_ld + colBlock + cq;
                const float* p1 = X + (size_t)(k0 + 2 * kp + 1) * x_ld + colBlock + cq;
                float4 a = *(const float4*)p0;
                float4 b = *(const float4*)p1;
                if (pre) {
                    a.x = fmaxf(a.x, 0.f); a.y = fmaxf(a.y, 0.f);
                    a.z = fmaxf(a.z, 0.f); a.w = fmaxf(a.w, 0.f);
                    b.x = fmaxf(b.x, 0.f); b.y = fmaxf(b.y, 0.f);
                    b.z = fmaxf(b.z, 0.f); b.w = fmaxf(b.w, 0.f);
                }
                uint4 wv;
                wv.x = pack2h(a.x, b.x); wv.y = pack2h(a.y, b.y);
                wv.z = pack2h(a.z, b.z); wv.w = pack2h(a.w, b.w);
                *(uint4*)&xtu[kp * TC + cq] = wv;
            }
        } else {
            for (int i = tid; i < KC * TC; i += 256) {
                const int k = i / TC, c = i % TC;
                const int kk = k0 + k, col = colBlock + c;
                float v = 0.0f;
                if (kk < C && col < M) v = X[(size_t)kk * x_ld + col];
                if (pre) v = fmaxf(v, 0.0f);
                xth[((k >> 1) * TC + c) * 2 + (k & 1)] = (_Float16)v;
            }
        }

        // prefetch next X K-chunk while this one is consumed
        if (kc + 1 < nK) {
            const int pk = tid >> 3;                  // 0..31
            if (k0 + KC + pk < C)
                __builtin_prefetch(X + (size_t)(k0 + KC + pk) * x_ld + colBlock + (tid & 7) * 32, 0, 1);
        }
        __syncthreads();

        // ---- fragment assembly per CDNA5 16-bit A/B layouts ----
        const int mr = lane & 15;
        const int gg = lane >> 4;
        v8u au, bu0, bu1;
        #pragma unroll
        for (int q = 0; q < 8; ++q) {
            const int ia = (q < 4) ? (gg * 4 + q) : (8 + gg * 4 + (q - 4));
            au[q]  = wtu[mr * (KC / 2) + ia];
            bu0[q] = xtu[ia * TC + wave * 32 + mr];
            bu1[q] = xtu[ia * TC + wave * 32 + 16 + mr];
        }
        const v16h afr = __builtin_bit_cast(v16h, au);
        acc0 = __builtin_amdgcn_wmma_f32_16x16x32_f16(
                   false, afr, false, __builtin_bit_cast(v16h, bu0), (short)0, acc0, false, false);
        acc1 = __builtin_amdgcn_wmma_f32_16x16x32_f16(
                   false, afr, false, __builtin_bit_cast(v16h, bu1), (short)0, acc1, false, false);
        __syncthreads();
    }

    // C/D layout: vgpr r -> M = r + 8*(lane>=16), N = lane&15
    const int ncol  = lane & 15;
    const int mbase = (lane >> 4) * 8;
    const int col0  = colBlock + wave * 32 + ncol;
    #pragma unroll
    for (int r = 0; r < 8; ++r) {
        const int o = oTile + r + mbase;
        if (o < O) {
            const float bv = bias ? bias[o] : 0.0f;
            #pragma unroll
            for (int t = 0; t < 2; ++t) {
                const int col = col0 + t * 16;
                if (col < M) {
                    float v = (t == 0 ? acc0[r] : acc1[r]) + bv;
                    const size_t oi = (size_t)o * out_ld + (size_t)col * cs + coff;
                    if (flags & FLAG_RES)  v += res[oi];
                    if (flags & FLAG_RELU) v = fmaxf(v, 0.0f);
                    if (flags & FLAG_TANH) v = tanhf(v);
                    out[oi] = v;
                }
            }
        }
    }
}

// ------------------------- small utility kernels ---------------------------

__global__ void zero_kernel(float* p, int n) {
    int i = blockIdx.x * blockDim.x + threadIdx.x;
    if (i < n) p[i] = 0.0f;
}

// (B,C,N) -> (C, B*N)
__global__ void transpose_bcn_kernel(const float* __restrict__ in, float* __restrict__ out,
                                     int Bn, int C, int Nn) {
    size_t i = (size_t)blockIdx.x * blockDim.x + threadIdx.x;
    size_t total = (size_t)Bn * C * Nn;
    if (i >= total) return;
    int n = (int)(i % Nn);
    size_t t = i / Nn;
    int c = (int)(t % C);
    int b = (int)(t / C);
    out[(size_t)c * ((size_t)Bn * Nn) + (size_t)b * Nn + n] = in[i];
}

// gmax[b*C + c] = max_n f[c, b*N + n]
// Stages the 2048-float row through LDS with CDNA5 async global->LDS copies
// (GLOBAL_LOAD_ASYNC_TO_LDS_B128, tracked by ASYNCcnt), then reduces from LDS.
__global__ __launch_bounds__(256)
void rowmax_kernel(const float* __restrict__ f, float* __restrict__ gmax,
                   int C, int Bn, int Nn) {
    int c = blockIdx.x, b = blockIdx.y;
    __shared__ float buf[NN];
    __shared__ float red[256];
    const float* src = f + (size_t)c * ((size_t)Bn * Nn) + (size_t)b * Nn;

    #pragma unroll
    for (int it = 0; it < NN / 1024; ++it) {
        const int e = it * 1024 + threadIdx.x * 4;       // 4 floats per lane per issue
        unsigned ldsoff = (unsigned)(uintptr_t)(&buf[e]);
        const float* g = src + e;
        asm volatile("global_load_async_to_lds_b128 %0, %1, off"
                     :: "v"(ldsoff), "v"(g) : "memory");
    }
    // each wave waits for its own async copies, then barrier makes LDS visible
    asm volatile("s_wait_asynccnt 0x0" ::: "memory");
    __syncthreads();

    float m = -INFINITY;
    for (int n = threadIdx.x; n < Nn; n += 256) m = fmaxf(m, buf[n]);
    red[threadIdx.x] = m;
    __syncthreads();
    for (int s = 128; s > 0; s >>= 1) {
        if (threadIdx.x < s) red[threadIdx.x] = fmaxf(red[threadIdx.x], red[threadIdx.x + s]);
        __syncthreads();
    }
    if (threadIdx.x == 0) gmax[b * C + c] = red[0];
}

// feat1 (258 x MM): [f128 ; gmax bcast ; feat_global bcast]
__global__ void feat1_kernel(const float* __restrict__ f, const float* __restrict__ gmax,
                             const float* __restrict__ fg, float* __restrict__ o) {
    size_t i = (size_t)blockIdx.x * blockDim.x + threadIdx.x;
    if (i >= (size_t)258 * MM) return;
    int c = (int)(i / MM), col = (int)(i % MM), b = col / NN;
    float v;
    if (c < 128)       v = f[(size_t)c * MM + col];
    else if (c < 256)  v = gmax[b * 128 + (c - 128)];
    else               v = fg[b * DIMF + (c - 256)];
    o[i] = v;
}

// o (C1+C2 x Mc) = [x1 ; x2]
__global__ void concat2_kernel(const float* __restrict__ x1, const float* __restrict__ x2,
                               float* __restrict__ o, int C1, int C2, int Mc) {
    size_t i = (size_t)blockIdx.x * blockDim.x + threadIdx.x;
    if (i >= (size_t)(C1 + C2) * Mc) return;
    int c = (int)(i / Mc), col = (int)(i % Mc);
    o[i] = (c < C1) ? x1[(size_t)c * Mc + col] : x2[(size_t)(c - C1) * Mc + col];
}

// kNN: one thread per (b,n) keeps a sorted top-16 of squared distances.
__global__ void knn_kernel(const float* __restrict__ pcdT, int* __restrict__ idx) {
    int t = blockIdx.x * blockDim.x + threadIdx.x;
    if (t >= BB * NN) return;
    int b = t / NN, n = t % NN;
    const float px = pcdT[0 * MM + b * NN + n];
    const float py = pcdT[1 * MM + b * NN + n];
    const float pz = pcdT[2 * MM + b * NN + n];
    float bd[KNNK]; int bi[KNNK];
    for (int k = 0; k < KNNK; ++k) { bd[k] = INFINITY; bi[k] = 0; }
    for (int m = 0; m < NN; ++m) {
        float dx = px - pcdT[0 * MM + b * NN + m];
        float dy = py - pcdT[1 * MM + b * NN + m];
        float dz = pz - pcdT[2 * MM + b * NN + m];
        float d = dx * dx + dy * dy + dz * dz;
        if (d < bd[KNNK - 1]) {
            int j = KNNK - 1;
            while (j > 0 && bd[j - 1] > d) { bd[j] = bd[j - 1]; bi[j] = bi[j - 1]; --j; }
            bd[j] = d; bi[j] = m;
        }
    }
    for (int k = 0; k < KNNK; ++k) idx[(size_t)t * KNNK + k] = bi[k];
}

// pos_rel (3 x BNK): p[n] - p[idx]
__global__ void posrel_kernel(const float* __restrict__ pcdT, const int* __restrict__ idx,
                              float* __restrict__ pr) {
    size_t col = (size_t)blockIdx.x * blockDim.x + threadIdx.x;
    if (col >= (size_t)BNK) return;
    int p = (int)(col >> 4);
    int b = p / NN, n = p % NN;
    int id = idx[col];
    for (int cc = 0; cc < 3; ++cc)
        pr[(size_t)cc * BNK + col] =
            pcdT[cc * MM + b * NN + n] - pcdT[cc * MM + b * NN + id];
}

// per-channel sum / sumsq accumulate (atomics into stats[c], stats[C+c])
__global__ void sumsq_kernel(const float* __restrict__ X, float* __restrict__ stats,
                             int C, int Mc) {
    int c = blockIdx.y;
    __shared__ float s1[256], s2[256];
    float a = 0.0f, q = 0.0f;
    for (int j = blockIdx.x * 256 + threadIdx.x; j < Mc; j += gridDim.x * 256) {
        float v = X[(size_t)c * Mc + j];
        a += v; q += v * v;
    }
    s1[threadIdx.x] = a; s2[threadIdx.x] = q;
    __syncthreads();
    for (int s = 128; s > 0; s >>= 1) {
        if (threadIdx.x < s) { s1[threadIdx.x] += s1[threadIdx.x + s];
                               s2[threadIdx.x] += s2[threadIdx.x + s]; }
        __syncthreads();
    }
    if (threadIdx.x == 0) { atomicAdd(&stats[c], s1[0]); atomicAdd(&stats[C + c], s2[0]); }
}

// BN finalize: A = g*rsqrt(var+eps); Bc = be - mean*A
__global__ void bnfin_kernel(const float* __restrict__ stats, const float* __restrict__ g,
                             const float* __restrict__ be, float* __restrict__ A,
                             float* __restrict__ Bc, int C, float invM) {
    int c = blockIdx.x * blockDim.x + threadIdx.x;
    if (c >= C) return;
    float mean = stats[c] * invM;
    float var  = stats[C + c] * invM - mean * mean;
    float a = g[c] * rsqrtf(var + 1e-5f);
    A[c] = a; Bc[c] = be[c] - mean * a;
}

// X = relu(X*A[c] + Bc[c]) in place
__global__ void affine_relu_kernel(float* __restrict__ X, const float* __restrict__ A,
                                   const float* __restrict__ Bc, int C, int Mc) {
    size_t i = (size_t)blockIdx.x * blockDim.x + threadIdx.x;
    if (i >= (size_t)C * Mc) return;
    int c = (int)(i / Mc);
    X[i] = fmaxf(X[i] * A[c] + Bc[c], 0.0f);
}

// attn_in[c,j] = qf[c,bn+n] - kf[c,bn+idx] + pe[c,j]   (chunk-local)
__global__ void attn_in_kernel(const float* __restrict__ qf, const float* __restrict__ kf,
                               const int* __restrict__ idx, const float* __restrict__ pe,
                               float* __restrict__ ai, int base) {
    size_t i = (size_t)blockIdx.x * blockDim.x + threadIdx.x;
    if (i >= (size_t)64 * CHCOL) return;
    int c = (int)(i / CHCOL), j = (int)(i % CHCOL);
    int gcol = base + j;
    int p = gcol >> 4;
    int b = p / NN, n = p % NN;
    int id = idx[gcol];
    ai[i] = qf[(size_t)c * MM + (size_t)b * NN + n]
          - kf[(size_t)c * MM + (size_t)b * NN + id] + pe[i];
}

// softmax over groups of 16 columns per channel row (chunk buffer 64 x CHCOL)
__global__ void softmax16_kernel(float* __restrict__ ab) {
    int i = blockIdx.x * blockDim.x + threadIdx.x;
    const int groups = CHCOL / KNNK;
    if (i >= 64 * groups) return;
    int c = i / groups, gidx = i % groups;
    float* p = ab + (size_t)c * CHCOL + (size_t)gidx * KNNK;
    float mx = -INFINITY;
    for (int k = 0; k < KNNK; ++k) mx = fmaxf(mx, p[k]);
    float e[KNNK], s = 0.0f;
    for (int k = 0; k < KNNK; ++k) { e[k] = expf(p[k] - mx); s += e[k]; }
    float inv = 1.0f / s;
    for (int k = 0; k < KNNK; ++k) p[k] = e[k] * inv;
}

// agg[c, p] = sum_k attn * (vf_gathered + pe)
__global__ void agg_kernel(const float* __restrict__ ab, const float* __restrict__ vf,
                           const float* __restrict__ pe, const int* __restrict__ idx,
                           float* __restrict__ agg, int basePts) {
    int i = blockIdx.x * blockDim.x + threadIdx.x;
    if (i >= 64 * CHPTS) return;
    int c = i / CHPTS, lp = i % CHPTS;
    int p = basePts + lp;
    int b = p / NN;
    float s = 0.0f;
    for (int k = 0; k < KNNK; ++k) {
        size_t j = (size_t)lp * KNNK + k;
        int id = idx[(size_t)p * KNNK + k];
        s += ab[(size_t)c * CHCOL + j]
           * (vf[(size_t)c * MM + (size_t)b * NN + id] + pe[(size_t)c * CHCOL + j]);
    }
    agg[(size_t)c * MM + p] = s;
}

// cat rows 128..255 = repeat(H, 2)
__global__ void hup_kernel(const float* __restrict__ H, float* __restrict__ cat) {
    size_t i = (size_t)blockIdx.x * blockDim.x + threadIdx.x;
    if (i >= (size_t)128 * MM2) return;
    int c = (int)(i / MM2), col2 = (int)(i % MM2);
    cat[(size_t)(128 + c) * MM2 + col2] = H[(size_t)c * MM + (col2 >> 1)];
}

// repack ps_deconv (32,128,2) into two (128 x 32) row-major weight matrices
__global__ void deconvw_kernel(const float* __restrict__ dw, float* __restrict__ w0,
                               float* __restrict__ w1) {
    int i = blockIdx.x * blockDim.x + threadIdx.x;
    if (i >= 32 * 128) return;
    int c = i / 128, d = i % 128;
    w0[d * 32 + c] = dw[c * 256 + d * 2 + 0];
    w1[d * 32 + c] = dw[c * 256 + d * 2 + 1];
}

// pcd_child (B,3,N2) = repeat(pcd_prev,2) + delta
__global__ void out_pcd_kernel(const float* __restrict__ pcd_prev,
                               const float* __restrict__ delta, float* __restrict__ out) {
    int i = blockIdx.x * blockDim.x + threadIdx.x;
    if (i >= BB * 3 * NN2) return;
    int n2 = i % NN2, c = (i / NN2) % 3, b = i / (3 * NN2);
    out[i] = pcd_prev[b * 3 * NN + c * NN + (n2 >> 1)]
           + delta[(size_t)c * MM2 + (size_t)b * NN2 + n2];
}

// K_curr (B,128,N2) appended after pcd_child
__global__ void out_K_kernel(const float* __restrict__ Kc, float* __restrict__ out) {
    size_t i = (size_t)blockIdx.x * blockDim.x + threadIdx.x;
    if (i >= (size_t)BB * 128 * NN2) return;
    int n2 = (int)(i % NN2), c = (int)((i / NN2) % 128), b = (int)(i / (128 * NN2));
    out[(size_t)(BB * 3 * NN2) + i] = Kc[(size_t)c * MM2 + (size_t)b * NN2 + n2];
}

// ------------------------------ host side ----------------------------------

static void gemm(hipStream_t s, const float* Wm, const float* X, const float* bias,
                 const float* res, float* out, int O, int C, int M, int x_ld,
                 int out_ld, int cs, int coff, int flags)
{
    dim3 grid((M + TC - 1) / TC, (O + 15) / 16);
    wmma_gemm_kernel<<<grid, 256, 0, s>>>(Wm, X, bias, res, out, O, C, M,
                                          x_ld, out_ld, cs, coff, flags);
}

static inline unsigned g1(size_t n) { return (unsigned)((n + 255) / 256); }

enum {
    IN_PCD = 0, IN_FG, IN_KPREV,
    P_MLP1_W1, P_MLP1_B1, P_MLP1_W2, P_MLP1_B2,
    P_MLP2_W1, P_MLP2_B1, P_MLP2_W2, P_MLP2_B2,
    P_V_W1, P_V_B1, P_V_W2, P_V_B2, P_V_WS, P_V_BS,
    P_K_W, P_K_B, P_Q_W, P_Q_B, P_VAL_W, P_VAL_B,
    P_POS_W1, P_POS_B1, P_POS_G, P_POS_BE, P_POS_W2, P_POS_B2,
    P_ATTN_W1, P_ATTN_B1, P_ATTN_G, P_ATTN_BE, P_ATTN_W2, P_ATTN_B2,
    P_END_W, P_END_B,
    P_PS_W1, P_PS_B1, P_PS_W2, P_PS_B2, P_PS_DECONV,
    P_DF_W1, P_DF_B1, P_DF_W2, P_DF_B2, P_DF_WS, P_DF_BS,
    P_D_W1, P_D_B1, P_D_W2, P_D_B2
};

extern "C" void kernel_launch(void* const* d_in, const int* in_sizes, int n_in,
                              void* d_out, int out_size, void* d_ws, size_t ws_size,
                              hipStream_t stream)
{
    (void)in_sizes; (void)n_in; (void)out_size; (void)ws_size;
    const float* pcd_prev = (const float*)d_in[IN_PCD];
    const float* feat_glb = (const float*)d_in[IN_FG];
    const float* K_prev   = (const float*)d_in[IN_KPREV];
    auto P = [&](int i) { return (const float*)d_in[i]; };
    float* ws  = (float*)d_ws;
    float* out = (float*)d_out;

    // workspace layout (floats), ~67M floats (~270 MB)
    size_t off = 0;
    auto AL = [&](size_t n) { size_t o = off; off += (n + 63) & ~(size_t)63; return o; };
    const size_t o_pcdT = AL((size_t)3 * MM),   o_KT   = AL((size_t)128 * MM);
    const size_t o_h64  = AL((size_t)64 * MM),  o_f128 = AL((size_t)128 * MM);
    const size_t o_gmax = AL(1024),             o_feat1 = AL((size_t)258 * MM);
    const size_t o_hid  = AL((size_t)256 * MM), o_Q    = AL((size_t)128 * MM);
    const size_t o_vin  = AL((size_t)256 * MM), o_vhid = AL((size_t)128 * MM);
    const size_t o_val  = AL((size_t)128 * MM);
    const size_t o_kf = AL((size_t)64 * MM), o_qf = AL((size_t)64 * MM), o_vf = AL((size_t)64 * MM);
    const size_t o_idx  = AL((size_t)BNK),      o_prel = AL((size_t)3 * BNK);
    const size_t o_stat = AL(512);
    const size_t o_Apos = AL(64), o_Bpos = AL(64), o_Aat = AL(256), o_Bat = AL(256);
    const size_t o_tA = AL((size_t)64 * CHCOL), o_pe = AL((size_t)64 * CHCOL);
    const size_t o_ai = AL((size_t)64 * CHCOL), o_tB = AL((size_t)256 * CHCOL);
    const size_t o_ab = AL((size_t)64 * CHCOL);
    const size_t o_agg = AL((size_t)64 * MM),   o_H   = AL((size_t)128 * MM);
    const size_t o_psh = AL((size_t)64 * MM),   o_fc  = AL((size_t)32 * MM);
    const size_t o_w0 = AL(128 * 32), o_w1 = AL(128 * 32);
    const size_t o_cat = AL((size_t)256 * MM2), o_dfh = AL((size_t)128 * MM2);
    const size_t o_Kc  = AL((size_t)128 * MM2), o_dh  = AL((size_t)64 * MM2);
    const size_t o_dlt = AL((size_t)3 * MM2);
    int* idxb = (int*)(ws + o_idx);

    // ---- layout: (B,C,N) -> (C, B*N) for the two batched inputs ----
    transpose_bcn_kernel<<<g1((size_t)BB * 3 * NN), 256, 0, stream>>>(pcd_prev, ws + o_pcdT, BB, 3, NN);
    transpose_bcn_kernel<<<g1((size_t)BB * 128 * NN), 256, 0, stream>>>(K_prev, ws + o_KT, BB, 128, NN);

    // ---- mlp_1 ----
    gemm(stream, P(P_MLP1_W1), ws + o_pcdT, P(P_MLP1_B1), nullptr, ws + o_h64,
         64, 3, MM, MM, MM, 1, 0, FLAG_RELU);
    gemm(stream, P(P_MLP1_W2), ws + o_h64, P(P_MLP1_B2), nullptr, ws + o_f128,
         128, 64, MM, MM, MM, 1, 0, 0);
    rowmax_kernel<<<dim3(128, BB), 256, 0, stream>>>(ws + o_f128, ws + o_gmax, 128, BB, NN);
    feat1_kernel<<<g1((size_t)258 * MM), 256, 0, stream>>>(ws + o_f128, ws + o_gmax, feat_glb, ws + o_feat1);

    // ---- mlp_2 -> Q ----
    gemm(stream, P(P_MLP2_W1), ws + o_feat1, P(P_MLP2_B1), nullptr, ws + o_hid,
         256, 258, MM, MM, MM, 1, 0, FLAG_RELU);
    gemm(stream, P(P_MLP2_W2), ws + o_hid, P(P_MLP2_B2), nullptr, ws + o_Q,
         128, 256, MM, MM, MM, 1, 0, 0);

    // ---- SkipTransformer value (MlpRes) ----
    concat2_kernel<<<g1((size_t)256 * MM), 256, 0, stream>>>(ws + o_KT, ws + o_Q, ws + o_vin, 128, 128, MM);
    gemm(stream, P(P_V_WS), ws + o_vin, P(P_V_BS), nullptr, ws + o_val,
         128, 256, MM, MM, MM, 1, 0, 0);                              // shortcut
    gemm(stream, P(P_V_W1), ws + o_vin, P(P_V_B1), nullptr, ws + o_vhid,
         128, 256, MM, MM, MM, 1, 0, FLAG_RELU);
    gemm(stream, P(P_V_W2), ws + o_vhid, P(P_V_B2), ws + o_val, ws + o_val,
         128, 128, MM, MM, MM, 1, 0, FLAG_RES);                       // value (= identity)

    // ---- k / q / v projections ----
    gemm(stream, P(P_K_W),   ws + o_KT,  P(P_K_B),   nullptr, ws + o_kf, 64, 128, MM, MM, MM, 1, 0, 0);
    gemm(stream, P(P_Q_W),   ws + o_Q,   P(P_Q_B),   nullptr, ws + o_qf, 64, 128, MM, MM, MM, 1, 0, 0);
    gemm(stream, P(P_VAL_W), ws + o_val, P(P_VAL_B), nullptr, ws + o_vf, 64, 128, MM, MM, MM, 1, 0, 0);

    // ---- kNN + pos_rel ----
    knn_kernel<<<g1(BB * NN), 256, 0, stream>>>(ws + o_pcdT, idxb);
    posrel_kernel<<<g1(BNK), 256, 0, stream>>>(ws + o_pcdT, idxb, ws + o_prel);

    const float invBNK = 1.0f / (float)BNK;

    // ---- pos-branch BN stats (chunked) ----
    zero_kernel<<<g1(512), 256, 0, stream>>>(ws + o_stat, 512);
    for (int ch = 0; ch < NCHNK; ++ch) {
        int base = ch * CHCOL;
        gemm(stream, P(P_POS_W1), ws + o_prel + base, P(P_POS_B1), nullptr, ws + o_tA,
             64, 3, CHCOL, BNK, CHCOL, 1, 0, 0);
        sumsq_kernel<<<dim3(32, 64), 256, 0, stream>>>(ws + o_tA, ws + o_stat, 64, CHCOL);
    }
    bnfin_kernel<<<1, 256, 0, stream>>>(ws + o_stat, P(P_POS_G), P(P_POS_BE),
                                        ws + o_Apos, ws + o_Bpos, 64, invBNK);

    // ---- attn-branch BN stats (chunked, recompute pe + attn_in) ----
    zero_kernel<<<g1(512), 256, 0, stream>>>(ws + o_stat, 512);
    for (int ch = 0; ch < NCHNK; ++ch) {
        int base = ch * CHCOL;
        gemm(stream, P(P_POS_W1), ws + o_prel + base, P(P_POS_B1), nullptr, ws + o_tA,
             64, 3, CHCOL, BNK, CHCOL, 1, 0, 0);
        affine_relu_kernel<<<g1((size_t)64 * CHCOL), 256, 0, stream>>>(ws + o_tA, ws + o_Apos, ws + o_Bpos, 64, CHCOL);
        gemm(stream, P(P_POS_W2), ws + o_tA, P(P_POS_B2), nullptr, ws + o_pe,
             64, 64, CHCOL, CHCOL, CHCOL, 1, 0, 0);
        attn_in_kernel<<<g1((size_t)64 * CHCOL), 256, 0, stream>>>(ws + o_qf, ws + o_kf, idxb, ws + o_pe, ws + o_ai, base);
        gemm(stream, P(P_ATTN_W1), ws + o_ai, P(P_ATTN_B1), nullptr, ws + o_tB,
             256, 64, CHCOL, CHCOL, CHCOL, 1, 0, 0);
        sumsq_kernel<<<dim3(32, 256), 256, 0, stream>>>(ws + o_tB, ws + o_stat, 256, CHCOL);
    }
    bnfin_kernel<<<1, 256, 0, stream>>>(ws + o_stat, P(P_ATTN_G), P(P_ATTN_BE),
                                        ws + o_Aat, ws + o_Bat, 256, invBNK);

    // ---- attention main pass (chunked): pe, attn, softmax, aggregate ----
    for (int ch = 0; ch < NCHNK; ++ch) {
        int base = ch * CHCOL;
        gemm(stream, P(P_POS_W1), ws + o_prel + base, P(P_POS_B1), nullptr, ws + o_tA,
             64, 3, CHCOL, BNK, CHCOL, 1, 0, 0);
        affine_relu_kernel<<<g1((size_t)64 * CHCOL), 256, 0, stream>>>(ws + o_tA, ws + o_Apos, ws + o_Bpos, 64, CHCOL);
        gemm(stream, P(P_POS_W2), ws + o_tA, P(P_POS_B2), nullptr, ws + o_pe,
             64, 64, CHCOL, CHCOL, CHCOL, 1, 0, 0);
        attn_in_kernel<<<g1((size_t)64 * CHCOL), 256, 0, stream>>>(ws + o_qf, ws + o_kf, idxb, ws + o_pe, ws + o_ai, base);
        gemm(stream, P(P_ATTN_W1), ws + o_ai, P(P_ATTN_B1), nullptr, ws + o_tB,
             256, 64, CHCOL, CHCOL, CHCOL, 1, 0, 0);
        affine_relu_kernel<<<g1((size_t)256 * CHCOL), 256, 0, stream>>>(ws + o_tB, ws + o_Aat, ws + o_Bat, 256, CHCOL);
        gemm(stream, P(P_ATTN_W2), ws + o_tB, P(P_ATTN_B2), nullptr, ws + o_ab,
             64, 256, CHCOL, CHCOL, CHCOL, 1, 0, 0);
        softmax16_kernel<<<g1((size_t)64 * (CHCOL / KNNK)), 256, 0, stream>>>(ws + o_ab);
        agg_kernel<<<g1((size_t)64 * CHPTS), 256, 0, stream>>>(ws + o_ab, ws + o_vf, ws + o_pe, idxb,
                                                               ws + o_agg, ch * CHPTS);
    }

    // ---- H = end_conv(agg) + identity ----
    gemm(stream, P(P_END_W), ws + o_agg, P(P_END_B), ws + o_val, ws + o_H,
         128, 64, MM, MM, MM, 1, 0, FLAG_RES);

    // ---- point splitting: mlp_ps + deconv ----
    gemm(stream, P(P_PS_W1), ws + o_H, P(P_PS_B1), nullptr, ws + o_psh,
         64, 128, MM, MM, MM, 1, 0, FLAG_RELU);
    gemm(stream, P(P_PS_W2), ws + o_psh, P(P_PS_B2), nullptr, ws + o_fc,
         32, 64, MM, MM, MM, 1, 0, 0);
    deconvw_kernel<<<g1(32 * 128), 256, 0, stream>>>(P(P_PS_DECONV), ws + o_w0, ws + o_w1);
    // deconv (ConvTranspose1d k=s=2) -> cat rows 0..127, strided column writes
    gemm(stream, ws + o_w0, ws + o_fc, nullptr, nullptr, ws + o_cat,
         128, 32, MM, MM, MM2, 2, 0, 0);
    gemm(stream, ws + o_w1, ws + o_fc, nullptr, nullptr, ws + o_cat,
         128, 32, MM, MM, MM2, 2, 1, 0);
    hup_kernel<<<g1((size_t)128 * MM2), 256, 0, stream>>>(ws + o_H, ws + o_cat);

    // ---- K_curr = MlpRes(cat) ----
    gemm(stream, P(P_DF_WS), ws + o_cat, P(P_DF_BS), nullptr, ws + o_Kc,
         128, 256, MM2, MM2, MM2, 1, 0, 0);
    gemm(stream, P(P_DF_W1), ws + o_cat, P(P_DF_B1), nullptr, ws + o_dfh,
         128, 256, MM2, MM2, MM2, 1, 0, FLAG_RELU);
    gemm(stream, P(P_DF_W2), ws + o_dfh, P(P_DF_B2), ws + o_Kc, ws + o_Kc,
         128, 128, MM2, MM2, MM2, 1, 0, FLAG_RES);

    // ---- delta = tanh(mlp_delta(relu(K_curr))) ----
    gemm(stream, P(P_D_W1), ws + o_Kc, P(P_D_B1), nullptr, ws + o_dh,
         64, 128, MM2, MM2, MM2, 1, 0, FLAG_RELU | FLAG_PRE_RELU);
    gemm(stream, P(P_D_W2), ws + o_dh, P(P_D_B2), nullptr, ws + o_dlt,
         3, 64, MM2, MM2, MM2, 1, 0, FLAG_TANH);

    // ---- outputs: pcd_child (B,3,N2) then K_curr (B,128,N2) ----
    out_pcd_kernel<<<g1((size_t)BB * 3 * NN2), 256, 0, stream>>>(pcd_prev, ws + o_dlt, out);
    out_K_kernel<<<g1((size_t)BB * 128 * NN2), 256, 0, stream>>>(ws + o_Kc, out);
}
